// CausalSelfAttention_59691455480636
// MI455X (gfx1250) — matrix operand order repaired and checked by reading, not compiled, not run on hardware
//
#include <hip/hip_runtime.h>

// ---------------- problem constants ----------------
#define B_  2
#define S_  2048
#define E_  2048
#define NH_ 16
#define NKV_ 4
#define HD_ 128
#define WINDOW_ 1024

typedef __attribute__((ext_vector_type(16))) __bf16 v16bf;
typedef __attribute__((ext_vector_type(8)))  float  v8f;

union ABu { v16bf v; unsigned int d[8]; };

__device__ __forceinline__ unsigned short f32_to_bf16(float f) {
  union { float f; unsigned int u; } c; c.f = f;
  unsigned int u = c.u;
  u += 0x7FFFu + ((u >> 16) & 1u);   // round-to-nearest-even
  return (unsigned short)(u >> 16);
}
__device__ __forceinline__ float bf16_to_f32(unsigned short h) {
  union { unsigned int u; float f; } c; c.u = ((unsigned int)h) << 16;
  return c.f;
}
__device__ __forceinline__ v8f wmma_bf16(v16bf a, v16bf b, v8f c) {
  return __builtin_amdgcn_wmma_f32_16x16x32_bf16(false, a, false, b, (short)0, c, false, false);
}

// ---- gfx1250 async global->LDS copy (ASYNCcnt path), inline asm ----
// Generic address of a __shared__ object is {aperture_hi32, lds_offset}
// (ISA 10.2: LDS aperture truncates addr[31:0]), so low 32 bits == LDS offset.
__device__ __forceinline__ unsigned lds_off_of(const void* p) {
  return (unsigned)(unsigned long long)p;
}
__device__ __forceinline__ void async_copy_b128(unsigned lds_off, const void* gptr) {
  asm volatile("global_load_async_to_lds_b128 %0, %1, off"
               :: "v"(lds_off), "v"((unsigned long long)(size_t)gptr)
               : "memory");
}
__device__ __forceinline__ void wait_async0() {
  asm volatile("s_wait_asynccnt 0" ::: "memory");
}

// ---------------- f32 -> bf16 convert (4 elems/thread) ----------------
__global__ void cvt_f32_bf16(const float* __restrict__ src,
                             unsigned short* __restrict__ dst, long n) {
  long i = ((long)blockIdx.x * blockDim.x + threadIdx.x) * 4;
  if (i + 3 >= n) { for (long j = i; j < n; ++j) dst[j] = f32_to_bf16(src[j]); return; }
  float4 f = *(const float4*)(src + i);
  unsigned int p0 = (unsigned int)f32_to_bf16(f.x) | ((unsigned int)f32_to_bf16(f.y) << 16);
  unsigned int p1 = (unsigned int)f32_to_bf16(f.z) | ((unsigned int)f32_to_bf16(f.w) << 16);
  *(uint2*)(dst + i) = make_uint2(p0, p1);
}

// ---------------- tiled WMMA GEMM: C[M,N] = A[M,K] * W[N,K]^T ----------------
// block tile 64(M) x 256(N), 8 waves (2x4), wave tile 32x64, K-step 32,
// double-buffered LDS fed by async global->LDS copies.
template <bool F32OUT>
__global__ __launch_bounds__(256)
void gemm_bf16(const unsigned short* __restrict__ A,
               const unsigned short* __restrict__ Bw,
               void* __restrict__ Cout, int M, int N, int K) {
  __shared__ unsigned short la[2][64][32];    // 2 x 4 KB
  __shared__ unsigned short lb[2][256][32];   // 2 x 16 KB

  const int tid  = threadIdx.x;
  const int lane = tid & 31;
  const int wave = tid >> 5;
  const int wm   = wave >> 2;     // 0..1  (M)
  const int wn   = wave & 3;      // 0..3  (N)
  const int hgrp = lane >> 4;
  const int l16  = lane & 15;

  const int n0 = blockIdx.x * 256;
  const int m0 = blockIdx.y * 64;

  v8f acc[2][4] = {};

  auto load_tiles = [&](int kk, int buf) {
    {   // A tile: 64x32 bf16 = 256 x b128, one per thread
      int row = tid >> 2, quad = tid & 3;
      async_copy_b128(lds_off_of(&la[buf][row][quad * 8]),
                      A + (size_t)(m0 + row) * K + kk + quad * 8);
    }
#pragma unroll
    for (int j = 0; j < 4; ++j) {  // B tile: 256x32 bf16 = 1024 x b128
      int q = tid + j * 256, row = q >> 2, quad = q & 3;
      async_copy_b128(lds_off_of(&lb[buf][row][quad * 8]),
                      Bw + (size_t)(n0 + row) * K + kk + quad * 8);
    }
  };

  load_tiles(0, 0);
  wait_async0();
  __syncthreads();

  int cur = 0;
  for (int kk = 0; kk < K; kk += 32) {
    if (kk + 32 < K) load_tiles(kk + 32, cur ^ 1);

    ABu af[2];
#pragma unroll
    for (int s = 0; s < 2; ++s) {
      const unsigned int* lau = (const unsigned int*)&la[cur][wm * 32 + s * 16 + l16][0];
#pragma unroll
      for (int j = 0; j < 8; ++j) {
        int kof = hgrp * 16 + j * 2;
        af[s].d[j] = lau[kof >> 1];
      }
    }
#pragma unroll
    for (int t = 0; t < 4; ++t) {
      ABu bf;
      const unsigned int* lbu = (const unsigned int*)&lb[cur][wn * 64 + t * 16 + l16][0];
#pragma unroll
      for (int j = 0; j < 8; ++j) bf.d[j] = lbu[hgrp * 8 + j];
      acc[0][t] = wmma_bf16(af[0].v, bf.v, acc[0][t]);
      acc[1][t] = wmma_bf16(af[1].v, bf.v, acc[1][t]);
    }

    wait_async0();
    __syncthreads();
    cur ^= 1;
  }

#pragma unroll
  for (int s = 0; s < 2; ++s)
#pragma unroll
    for (int t = 0; t < 4; ++t) {
      int col = n0 + wn * 64 + t * 16 + l16;
#pragma unroll
      for (int r = 0; r < 8; ++r) {
        int row = m0 + wm * 32 + s * 16 + r + hgrp * 8;
        float v = acc[s][t][r];
        if constexpr (F32OUT) ((float*)Cout)[(size_t)row * N + col] = v;
        else ((unsigned short*)Cout)[(size_t)row * N + col] = f32_to_bf16(v);
      }
    }
}

// ---------------- RoPE on bf16 q/k (pair d, d+HD/2) ----------------
__global__ void rope_kernel(unsigned short* __restrict__ t,
                            const float* __restrict__ cosT,
                            const float* __restrict__ sinT,
                            int nheads, long total) {
  long idx = (long)blockIdx.x * blockDim.x + threadIdx.x;
  if (idx >= total) return;
  int d = (int)(idx & 63);
  int h = (int)((idx >> 6) % nheads);
  int s = (int)((idx / (64L * nheads)) % S_);
  int b = (int)(idx / (64L * nheads * S_));
  size_t base = (((size_t)b * S_ + s) * nheads + h) * HD_;
  float x1 = bf16_to_f32(t[base + d]);
  float x2 = bf16_to_f32(t[base + d + 64]);
  float c1 = cosT[(size_t)s * HD_ + d];
  float s1 = sinT[(size_t)s * HD_ + d];
  float c2 = cosT[(size_t)s * HD_ + d + 64];
  float s2 = sinT[(size_t)s * HD_ + d + 64];
  t[base + d]      = f32_to_bf16(x1 * c1 - x2 * s1);
  t[base + d + 64] = f32_to_bf16(x2 * c2 + x1 * s2);
}

// ---------------- windowed-causal GQA flash attention ----------------
// grid: B*NH*(S/64) blocks, 128 threads (4 waves); wave w owns 16 query rows.
__global__ __launch_bounds__(128)
void flash_attn(const unsigned short* __restrict__ Q,
                const unsigned short* __restrict__ Kt,
                const unsigned short* __restrict__ Vt,
                unsigned short* __restrict__ O) {
  __shared__ unsigned short lk[32][136];    // K chunk [key][hd], padded (272B rows)
  __shared__ unsigned short lvt[128][36];   // V chunk transposed [hd][key]
  __shared__ unsigned short lp[4][16][32];  // per-wave P tile (C->A transpose)

  const int tid  = threadIdx.x;
  const int lane = tid & 31;
  const int wave = tid >> 5;
  const int hgrp = lane >> 4;
  const int l16  = lane & 15;

  const int nqb = S_ / 64;
  const int q0  = (blockIdx.x % nqb) * 64;
  const int h   = (blockIdx.x / nqb) % NH_;
  const int b   = blockIdx.x / (nqb * NH_);
  const int kvh = h / (NH_ / NKV_);
  const int qr  = q0 + wave * 16;

  // Q fragments (A-layout), 4 chunks of K-dim 32
  ABu qf[4];
  {
    const unsigned short* qrow = Q + (((size_t)b * S_ + qr + l16) * NH_ + h) * HD_;
#pragma unroll
    for (int kk = 0; kk < 4; ++kk)
#pragma unroll
      for (int j = 0; j < 8; ++j) {
        int kof = kk * 32 + hgrp * 16 + j * 2;
        qf[kk].d[j] = *(const unsigned int*)(qrow + kof);
      }
  }

  v8f o_acc[8] = {};
  float row_max[8], row_sum[8];
#pragma unroll
  for (int r = 0; r < 8; ++r) { row_max[r] = -3.0e38f; row_sum[r] = 0.0f; }
  const float scale = 0.08838834764831845f;  // 1/sqrt(128)

  int kstart = q0 - (WINDOW_ - 1); if (kstart < 0) kstart = 0; kstart &= ~31;
  const int kend = q0 + 64;

  for (int c0 = kstart; c0 < kend; c0 += 32) {
    __syncthreads();
    // stage K chunk via async global->LDS: 32 keys x 128 hd bf16 = 512 x b128
#pragma unroll
    for (int j = 0; j < 4; ++j) {
      int q = tid + j * 128, row = q >> 4, quad = q & 15;
      async_copy_b128(lds_off_of(&lk[row][quad * 8]),
                      Kt + (((size_t)b * S_ + c0 + row) * NKV_ + kvh) * HD_ + quad * 8);
    }
    // stage V chunk transposed (manual: async copies cannot transpose)
#pragma unroll
    for (int j = 0; j < 16; ++j) {
      int idx = tid + j * 128;
      int key = idx >> 6, hp = idx & 63;
      unsigned int uv = *(const unsigned int*)(Vt + (((size_t)b * S_ + c0 + key) * NKV_ + kvh) * HD_ + hp * 2);
      lvt[hp * 2][key]     = (unsigned short)(uv & 0xFFFFu);
      lvt[hp * 2 + 1][key] = (unsigned short)(uv >> 16);
    }
    wait_async0();
    __syncthreads();

    // scores: S[16q x 32k] = Q * K^T  (2 n-tiles x 4 depth-chunks = 8 WMMAs)
    v8f s0 = {}, s1 = {};
#pragma unroll
    for (int kk = 0; kk < 4; ++kk) {
      ABu bf0, bf1;
      const unsigned int* r0 = (const unsigned int*)&lk[l16][0];
      const unsigned int* r1 = (const unsigned int*)&lk[16 + l16][0];
#pragma unroll
      for (int j = 0; j < 8; ++j) {
        int dw = kk * 16 + hgrp * 8 + j;
        bf0.d[j] = r0[dw];
        bf1.d[j] = r1[dw];
      }
      s0 = wmma_bf16(qf[kk].v, bf0.v, s0);
      s1 = wmma_bf16(qf[kk].v, bf1.v, s1);
    }

    // scale + window mask + online softmax (row spread over 16 lanes of a half)
    float p0v[8], p1v[8];
#pragma unroll
    for (int r = 0; r < 8; ++r) {
      int qi = qr + r + hgrp * 8;
      int j0 = c0 + l16, j1 = j0 + 16;
      float a0 = s0[r] * scale, a1 = s1[r] * scale;
      int d0 = qi - j0, d1 = qi - j1;
      if (!(d0 >= 0 && d0 < WINDOW_)) a0 = -1.0e30f;
      if (!(d1 >= 0 && d1 < WINDOW_)) a1 = -1.0e30f;
      float m = fmaxf(a0, a1);
#pragma unroll
      for (int off = 1; off < 16; off <<= 1) m = fmaxf(m, __shfl_xor(m, off, 32));
      float nm = fmaxf(row_max[r], m);
      float alpha = __expf(row_max[r] - nm);
      row_max[r] = nm;
      float e0 = __expf(a0 - nm), e1 = __expf(a1 - nm);
      float sum = e0 + e1;
#pragma unroll
      for (int off = 1; off < 16; off <<= 1) sum += __shfl_xor(sum, off, 32);
      row_sum[r] = row_sum[r] * alpha + sum;
#pragma unroll
      for (int t = 0; t < 8; ++t) o_acc[t][r] *= alpha;
      p0v[r] = e0; p1v[r] = e1;
    }

    // C-layout -> A-layout for P via per-wave LDS tile
#pragma unroll
    for (int r = 0; r < 8; ++r) {
      int rl = r + hgrp * 8;
      lp[wave][rl][l16]      = f32_to_bf16(p0v[r]);
      lp[wave][rl][16 + l16] = f32_to_bf16(p1v[r]);
    }
    ABu pf;
    const unsigned int* pr = (const unsigned int*)&lp[wave][l16][0];
#pragma unroll
    for (int j = 0; j < 8; ++j) {
      int kof = hgrp * 16 + j * 2;
      pf.d[j] = pr[kof >> 1];
    }

    // O[16 x 128] += P[16 x 32] * V[32 x 128]  (8 WMMAs)
#pragma unroll
    for (int t = 0; t < 8; ++t) {
      ABu vf;
      const unsigned int* vr = (const unsigned int*)&lvt[t * 16 + l16][0];
#pragma unroll
      for (int j = 0; j < 8; ++j) vf.d[j] = vr[hgrp * 8 + j];
      o_acc[t] = wmma_bf16(pf.v, vf.v, o_acc[t]);
    }
  }

  // normalize + store bf16 into [B*S, NH*HD]
#pragma unroll
  for (int t = 0; t < 8; ++t) {
    int col = h * HD_ + t * 16 + l16;
#pragma unroll
    for (int r = 0; r < 8; ++r) {
      int qi = qr + r + hgrp * 8;
      float v = o_acc[t][r] / row_sum[r];
      O[((size_t)b * S_ + qi) * E_ + col] = f32_to_bf16(v);
    }
  }
}

// ---------------- workspace layout (bytes) ----------------
static constexpr size_t OFF_XB  = 0;                    // x bf16       16 MB
static constexpr size_t OFF_WQB = 16777216;             // Wq bf16       8 MB
static constexpr size_t OFF_WKB = 25165824;             // Wk bf16       2 MB
static constexpr size_t OFF_WVB = 27262976;             // Wv bf16       2 MB
static constexpr size_t OFF_WOB = 29360128;             // Wo bf16       8 MB
static constexpr size_t OFF_Q   = 37748736;             // q bf16       16 MB
static constexpr size_t OFF_K   = 54525952;             // k bf16        4 MB
static constexpr size_t OFF_V   = 58720256;             // v bf16        4 MB
static constexpr size_t OFF_AO  = 62914560;             // attn out bf16 16 MB

extern "C" void kernel_launch(void* const* d_in, const int* in_sizes, int n_in,
                              void* d_out, int out_size, void* d_ws, size_t ws_size,
                              hipStream_t stream) {
  (void)in_sizes; (void)n_in; (void)out_size; (void)ws_size;
  const float* x    = (const float*)d_in[0];
  const float* cosT = (const float*)d_in[1];
  const float* sinT = (const float*)d_in[2];
  const float* Wq   = (const float*)d_in[3];
  const float* Wk   = (const float*)d_in[4];
  const float* Wv   = (const float*)d_in[5];
  const float* Wo   = (const float*)d_in[6];

  char* ws = (char*)d_ws;
  unsigned short* XB  = (unsigned short*)(ws + OFF_XB);
  unsigned short* WQB = (unsigned short*)(ws + OFF_WQB);
  unsigned short* WKB = (unsigned short*)(ws + OFF_WKB);
  unsigned short* WVB = (unsigned short*)(ws + OFF_WVB);
  unsigned short* WOB = (unsigned short*)(ws + OFF_WOB);
  unsigned short* QB  = (unsigned short*)(ws + OFF_Q);
  unsigned short* KB  = (unsigned short*)(ws + OFF_K);
  unsigned short* VB  = (unsigned short*)(ws + OFF_V);
  unsigned short* AO  = (unsigned short*)(ws + OFF_AO);

  const long nX  = (long)B_ * S_ * E_;        // 8388608
  const long nWq = (long)NH_ * HD_ * E_;      // 4194304
  const long nWk = (long)NKV_ * HD_ * E_;     // 1048576
  const long nWo = (long)E_ * E_;             // 4194304

  cvt_f32_bf16<<<(unsigned)(nX  / 1024), 256, 0, stream>>>(x,  XB,  nX);
  cvt_f32_bf16<<<(unsigned)(nWq / 1024), 256, 0, stream>>>(Wq, WQB, nWq);
  cvt_f32_bf16<<<(unsigned)(nWk / 1024), 256, 0, stream>>>(Wk, WKB, nWk);
  cvt_f32_bf16<<<(unsigned)(nWk / 1024), 256, 0, stream>>>(Wv, WVB, nWk);
  cvt_f32_bf16<<<(unsigned)(nWo / 1024), 256, 0, stream>>>(Wo, WOB, nWo);

  const int M = B_ * S_;  // 4096
  // Q = XB @ Wq^T  [4096 x 2048]
  gemm_bf16<false><<<dim3(2048 / 256, M / 64), 256, 0, stream>>>(XB, WQB, QB, M, 2048, E_);
  // K,V = XB @ Wk^T / Wv^T  [4096 x 512]
  gemm_bf16<false><<<dim3(512 / 256, M / 64), 256, 0, stream>>>(XB, WKB, KB, M, 512, E_);
  gemm_bf16<false><<<dim3(512 / 256, M / 64), 256, 0, stream>>>(XB, WVB, VB, M, 512, E_);

  // RoPE
  const long nQr = (long)B_ * S_ * NH_ * 64;   // 4194304
  const long nKr = (long)B_ * S_ * NKV_ * 64;  // 1048576
  rope_kernel<<<(unsigned)(nQr / 256), 256, 0, stream>>>(QB, cosT, sinT, NH_, nQr);
  rope_kernel<<<(unsigned)(nKr / 256), 256, 0, stream>>>(KB, cosT, sinT, NKV_, nKr);

  // flash attention
  flash_attn<<<B_ * NH_ * (S_ / 64), 128, 0, stream>>>(QB, KB, VB, AO);

  // out = AO @ Wo^T -> f32 d_out
  gemm_bf16<true><<<dim3(2048 / 256, M / 64), 256, 0, stream>>>(AO, WOB, d_out, M, 2048, E_);
}
